// RNN_14087492731481
// MI455X (gfx1250) — matrix operand order repaired and checked by reading, not compiled, not run on hardware
//
#include <hip/hip_runtime.h>

// ---------------------------------------------------------------------------
// RNN forward for MI455X (gfx1250, wave32, WMMA).
//   phase 0: transpose+convert weights to bf16 [N][K] layout (for B fragments)
//   phase 1: xh = bf16(x @ W_ih + b_ih)          (parallel GEMM, WMMA bf16)
//   phase 2: h_t = tanh(xh_t + h_{t-1} @ W_hh)   (serial scan, 4 WGs, batch-split,
//            W_hh fragments register-resident, async global->LDS xh prefetch)
//   phase 3: y = hs @ W_ho + b_ho                (parallel GEMM, WMMA bf16)
// ---------------------------------------------------------------------------

typedef __attribute__((ext_vector_type(16))) __bf16 v16bf;
typedef __attribute__((ext_vector_type(8)))  __bf16 v8bf;
typedef __attribute__((ext_vector_type(4)))  __bf16 v4bf;
typedef __attribute__((ext_vector_type(8)))  float  v8f;
typedef __attribute__((ext_vector_type(4)))  float  v4f;
typedef __attribute__((ext_vector_type(4)))  int    v4i;

#define NT 4096   // timesteps
#define NB 64     // batch
#define NH 256    // I == H == O

#if defined(__AMDGCN__) && \
    __has_builtin(__builtin_amdgcn_global_load_async_to_lds_b128) && \
    __has_builtin(__builtin_amdgcn_s_wait_asynccnt)
#define SCAN_ASYNC 1
// builtin prototype (from hipcc diagnostic): (v4i AS1*, v4i AS3*, imm, imm)
typedef __attribute__((address_space(1))) v4i* gv4i_p;
typedef __attribute__((address_space(3))) v4i* lv4i_p;
#else
#define SCAN_ASYNC 0
#endif

static __device__ __forceinline__ v8f wmma_bf16(v16bf a, v16bf b, v8f c) {
  // D = A(16x32 bf16) * B(32x16 bf16) + C(16x16 f32)
  return __builtin_amdgcn_wmma_f32_16x16x32_bf16(false, a, false, b,
                                                 (short)0, c, false, false);
}

static __device__ __forceinline__ v16bf cat8(v8bf lo, v8bf hi) {
  return __builtin_shufflevector(lo, hi, 0, 1, 2, 3, 4, 5, 6, 7,
                                         8, 9, 10, 11, 12, 13, 14, 15);
}

// A fragment: 16x32 bf16. Lanes 0-15 hold K kb..kb+15,
// lanes 16-31 hold K kb+16..kb+31 (same K order as b_frag); M = lane%16.
static __device__ __forceinline__ v16bf a_frag(const __bf16* As, int row, int kb, int half) {
  const __bf16* p = As + row * NH + kb + half * 16;
  return cat8(*(const v8bf*)p, *(const v8bf*)(p + 8));
}

// B fragment: 32x16 bf16 from W^T stored [N][K]. Lanes 0-15: N=lane, K=kb..kb+15;
// lanes 16-31: N=lane-16, K=kb+16..kb+31 -> 16 contiguous values per lane.
static __device__ __forceinline__ v16bf b_frag(const __bf16* Wt, int n, int kb, int half) {
  const __bf16* p = Wt + n * NH + kb + half * 16;
  return cat8(*(const v8bf*)p, *(const v8bf*)(p + 8));
}

static __device__ __forceinline__ v4bf cvt4(v4f v) {
  v4bf b;
  b.x = (__bf16)v.x; b.y = (__bf16)v.y; b.z = (__bf16)v.z; b.w = (__bf16)v.w;
  return b;
}

// ---------------------------------------------------------------------------
// phase 0: Wt[n][k] = bf16(W[k][n])   (256x256)
// ---------------------------------------------------------------------------
__global__ __launch_bounds__(256) void k_wcvt(const float* __restrict__ W,
                                              __bf16* __restrict__ Wt) {
  const int idx = blockIdx.x * 256 + threadIdx.x;   // 65536 elements
  const int k = idx >> 8, n = idx & 255;
  Wt[n * NH + k] = (__bf16)W[k * NH + n];
}

// ---------------------------------------------------------------------------
// phase 1: xh(bf16)[M=T*B, 256] = x(f32) @ W_ih + b_ih
// 256 threads = 8 waves; WG tile 128(M) x 256(N); wave: 16(M) x 256(N).
// LDS: W_ih^T bf16 128KB + A tile bf16 64KB = 192KB.
// ---------------------------------------------------------------------------
__global__ __launch_bounds__(256) void k_gemm_xh(const float*  __restrict__ x,
                                                 const __bf16* __restrict__ WihT,
                                                 const float*  __restrict__ bih,
                                                 __bf16*       __restrict__ xh) {
  extern __shared__ char smem[];
  __bf16* Ws = (__bf16*)smem;               // [256][256]
  __bf16* As = (__bf16*)(smem + 131072);    // [128][256]
  const int  tid   = threadIdx.x;
  const long mbase = (long)blockIdx.x * 128;

  { // stage W^T (bf16 copy, 8192 x 16B)
    const v8bf* s = (const v8bf*)WihT;
    v8bf*       d = (v8bf*)Ws;
    #pragma unroll
    for (int i = 0; i < 32; ++i) d[tid + 256 * i] = s[tid + 256 * i];
  }
  { // stage A with f32->bf16 conversion (8192 float4 coalesced)
    const v4f* s = (const v4f*)(x + mbase * NH);
    #pragma unroll
    for (int i = 0; i < 32; ++i) {
      const int c = tid + 256 * i;
      *(v4bf*)(As + c * 4) = cvt4(s[c]);
    }
  }
  __syncthreads();

  const int wave = tid >> 5, lane = tid & 31, half = lane >> 4, r = lane & 15;
  const int arow = wave * 16 + r;

  v8f acc[16];
  #pragma unroll
  for (int i = 0; i < 16; ++i)
    #pragma unroll
    for (int j = 0; j < 8; ++j) acc[i][j] = 0.0f;

  #pragma unroll
  for (int k8 = 0; k8 < 8; ++k8) {
    const int kb = k8 * 32;
    const v16bf a = a_frag(As, arow, kb, half);
    #pragma unroll
    for (int nt = 0; nt < 16; ++nt) {
      const v16bf b = b_frag(Ws, nt * 16 + r, kb, half);
      acc[nt] = wmma_bf16(a, b, acc[nt]);
    }
  }

  #pragma unroll
  for (int nt = 0; nt < 16; ++nt) {
    const int   col  = nt * 16 + r;
    const float bias = bih[col];
    #pragma unroll
    for (int v = 0; v < 8; ++v) {
      const long row = mbase + wave * 16 + v + 8 * half;
      xh[row * NH + col] = (__bf16)(acc[nt][v] + bias);
    }
  }
}

// ---------------------------------------------------------------------------
// phase 2: serial scan. 4 workgroups, each owns 16 independent batch rows.
// 512 threads = 16 waves, wave w owns N-tile w.
// W_hh^T B-fragments are loaded ONCE into registers (64 VGPRs/lane) and stay
// resident for all 4096 steps: per step only A-fragment ds_loads remain.
// xh tiles (bf16, 8KB) double-buffered; prefetch of t+1 goes straight
// global->LDS via the CDNA5 async path (ASYNCcnt) when available.
// LDS: h 8KB + 2 x 8KB xh tiles = 24KB.
// ---------------------------------------------------------------------------
__global__ __launch_bounds__(512) void k_scan(const __bf16* __restrict__ xh,
                                              const float*  __restrict__ h0,
                                              const __bf16* __restrict__ WhhT,
                                              __bf16*       __restrict__ hs) {
  extern __shared__ char smem[];
  __bf16* Hs  = (__bf16*)smem;               // [16][256] 8KB
  __bf16* Xb0 = (__bf16*)(smem + 8192);      // [16][256] 8KB
  __bf16* Xb1 = (__bf16*)(smem + 16384);     // [16][256] 8KB
  const int tid   = threadIdx.x;
  const int bbase = blockIdx.x * 16;

  const int wave = tid >> 5, lane = tid & 31, half = lane >> 4, r = lane & 15;
  const int col = wave * 16 + r;

  // W_hh^T B-fragments -> registers for the entire scan (one-time global reads)
  v16bf bw[8];
  #pragma unroll
  for (int k8 = 0; k8 < 8; ++k8) bw[k8] = b_frag(WhhT, col, k8 * 32, half);

  { // stage h0 rows (f32 -> bf16)
    const v4f* s = (const v4f*)(h0 + bbase * NH);
    #pragma unroll
    for (int i = 0; i < 2; ++i) {
      const int c = tid + 512 * i;
      *(v4bf*)(Hs + c * 4) = cvt4(s[c]);
    }
  }
  { // stage xh tile for t=0
    ((v8bf*)Xb0)[tid] = ((const v8bf*)(xh + (long)bbase * NH))[tid];
  }
  __syncthreads();

  for (int t = 0; t < NT; ++t) {
    __bf16* xcur = (t & 1) ? Xb1 : Xb0;
    __bf16* xnxt = (t & 1) ? Xb0 : Xb1;

    const int tn = (t + 1 < NT) ? t + 1 : t;
    const __bf16* gsrc = xh + ((long)tn * NB + bbase) * NH + tid * 8;

#if SCAN_ASYNC
    // async global -> LDS copy of next xh tile (one b128 chunk per thread)
    __builtin_amdgcn_global_load_async_to_lds_b128(
        (gv4i_p)(void*)gsrc,
        (lv4i_p)(void*)(xnxt + tid * 8),
        0, 0);
#else
    const v8bf pf = *(const v8bf*)gsrc;
    const int tp = (t + 2 < NT) ? t + 2 : t;
    __builtin_prefetch(xh + ((long)tp * NB + bbase) * NH + tid * 8, 0, 1);
#endif

    // issue all A-fragment LDS loads for this step together (one wait, not 8)
    v16bf af[8];
    #pragma unroll
    for (int k8 = 0; k8 < 8; ++k8) af[k8] = a_frag(Hs, r, k8 * 32, half);

    v8f acc;
    #pragma unroll
    for (int j = 0; j < 8; ++j) acc[j] = 0.0f;
    #pragma unroll
    for (int k8 = 0; k8 < 8; ++k8) acc = wmma_bf16(af[k8], bw[k8], acc);

    float hv[8];
    #pragma unroll
    for (int v = 0; v < 8; ++v) {
      const int row = v + 8 * half;
      hv[v] = tanhf((float)xcur[row * NH + col] + acc[v]);
    }
    __syncthreads();  // all reads of Hs / xcur complete

    const long g = ((long)t * NB + bbase) * NH;
    #pragma unroll
    for (int v = 0; v < 8; ++v) {
      const int    row = v + 8 * half;
      const __bf16 hb  = (__bf16)hv[v];
      Hs[row * NH + col]     = hb;   // h for next step
      hs[g + row * NH + col] = hb;   // record for output GEMM
    }
#if SCAN_ASYNC
    __builtin_amdgcn_s_wait_asynccnt(0);   // next tile resident in LDS
#else
    *(v8bf*)(xnxt + tid * 8) = pf;
#endif
    __syncthreads();
  }
}

// ---------------------------------------------------------------------------
// phase 3: y(f32) = hs(bf16) @ W_ho + b_ho.  Same tiling as phase 1.
// ---------------------------------------------------------------------------
__global__ __launch_bounds__(256) void k_gemm_out(const __bf16* __restrict__ hsm,
                                                  const __bf16* __restrict__ WhoT,
                                                  const float*  __restrict__ bho,
                                                  float*        __restrict__ y) {
  extern __shared__ char smem[];
  __bf16* Ws = (__bf16*)smem;               // 128KB
  __bf16* As = (__bf16*)(smem + 131072);    // 64KB
  const int  tid   = threadIdx.x;
  const long mbase = (long)blockIdx.x * 128;

  { const v8bf* s = (const v8bf*)WhoT;
    v8bf*       d = (v8bf*)Ws;
    #pragma unroll
    for (int i = 0; i < 32; ++i) d[tid + 256 * i] = s[tid + 256 * i];
  }
  { const v8bf* s = (const v8bf*)(hsm + mbase * NH);
    v8bf*       d = (v8bf*)As;
    #pragma unroll
    for (int i = 0; i < 16; ++i) d[tid + 256 * i] = s[tid + 256 * i];
  }
  __syncthreads();

  const int wave = tid >> 5, lane = tid & 31, half = lane >> 4, r = lane & 15;
  const int arow = wave * 16 + r;

  v8f acc[16];
  #pragma unroll
  for (int i = 0; i < 16; ++i)
    #pragma unroll
    for (int j = 0; j < 8; ++j) acc[i][j] = 0.0f;

  #pragma unroll
  for (int k8 = 0; k8 < 8; ++k8) {
    const int kb = k8 * 32;
    const v16bf a = a_frag(As, arow, kb, half);
    #pragma unroll
    for (int nt = 0; nt < 16; ++nt) {
      const v16bf b = b_frag(Ws, nt * 16 + r, kb, half);
      acc[nt] = wmma_bf16(a, b, acc[nt]);
    }
  }

  #pragma unroll
  for (int nt = 0; nt < 16; ++nt) {
    const int   col  = nt * 16 + r;
    const float bias = bho[col];
    #pragma unroll
    for (int v = 0; v < 8; ++v) {
      const long row = mbase + wave * 16 + v + 8 * half;
      y[row * NH + col] = acc[nt][v] + bias;
    }
  }
}

// ---------------------------------------------------------------------------
extern "C" void kernel_launch(void* const* d_in, const int* in_sizes, int n_in,
                              void* d_out, int out_size, void* d_ws, size_t ws_size,
                              hipStream_t stream) {
  (void)in_sizes; (void)n_in; (void)out_size; (void)ws_size;

  const float* x   = (const float*)d_in[0];   // [4096,64,256]
  const float* h0  = (const float*)d_in[1];   // [64,256]
  const float* Wih = (const float*)d_in[2];   // [256,256]
  const float* bih = (const float*)d_in[3];   // [256]
  const float* Whh = (const float*)d_in[4];   // [256,256]
  const float* Who = (const float*)d_in[5];   // [256,256]
  const float* bho = (const float*)d_in[6];   // [256]
  float* y = (float*)d_out;                   // [4096,64,256] f32

  // workspace layout (total 268,828,672 bytes ~= 257 MB)
  char*   ws   = (char*)d_ws;
  __bf16* WihT = (__bf16*)(ws + 0);                        // 128KB
  __bf16* WhhT = (__bf16*)(ws + 131072);                   // 128KB
  __bf16* WhoT = (__bf16*)(ws + 262144);                   // 128KB
  __bf16* xh   = (__bf16*)(ws + 393216);                   // 134,217,728 B
  __bf16* hs   = (__bf16*)(ws + 393216 + 134217728);       // 134,217,728 B

  k_wcvt<<<256, 256, 0, stream>>>(Wih, WihT);
  k_wcvt<<<256, 256, 0, stream>>>(Whh, WhhT);
  k_wcvt<<<256, 256, 0, stream>>>(Who, WhoT);

  // M = T*B = 262144 rows, 128 rows per workgroup
  k_gemm_xh <<<2048, 256, 196608, stream>>>(x, WihT, bih, xh);
  k_scan    <<<   4, 512,  24576, stream>>>(xh, h0, WhhT, hs);
  k_gemm_out<<<2048, 256, 196608, stream>>>(hs, WhoT, bho, y);
}
